// TemporalInitial_76862734729471
// MI455X (gfx1250) — matrix operand order repaired and checked by reading, not compiled, not run on hardware
//
#include <hip/hip_runtime.h>
#include <hip/hip_bf16.h>
#include <stdint.h>

// Problem sizes
#define B_    128
#define T_    2048
#define DIN   128
#define HID_  132
#define DOUT  64
#define M_    (B_ * T_)      // 262144 rows
#define NC    16             // time chunks for suffix-max scan
#define TC    (T_ / NC)      // 128

// Layout constants
#define SA    168            // activation row stride in bf16 elements (K padded to 160, +8 pad)

// Fragment-major weight storage sizes (dwords): KT * NT * 32 lanes * 8 dwords
#define W1_DW  (4 * 9 * 32 * 8)   // 9216
#define W23_DW (5 * 9 * 32 * 8)   // 11520 (shared by W2 and W3, refilled between layers)
#define W4_DW  (5 * 4 * 32 * 8)   // 5120

typedef __bf16 bf16_t;
typedef bf16_t v16bf __attribute__((ext_vector_type(16)));
typedef float  v8f   __attribute__((ext_vector_type(8)));

union Frag { v16bf v; uint4 q[2]; uint32_t u[8]; };

__device__ __forceinline__ uint16_t f2bf(float f) {
    uint32_t u = __float_as_uint(f);
    u += 0x7fffu + ((u >> 16) & 1u);   // round-to-nearest-even
    return (uint16_t)(u >> 16);
}

// ---------------- suffix cummax (3-pass chunked scan) ----------------

__global__ void k_chunkmax(const float* __restrict__ P, float* __restrict__ cmax) {
    const int b = blockIdx.x, c = blockIdx.y, d = threadIdx.x;
    const float* p = P + ((size_t)b * T_ + (size_t)c * TC) * DIN + d;
    float m = -INFINITY;
    for (int i = 0; i < TC; ++i) m = fmaxf(m, p[(size_t)i * DIN]);
    cmax[((size_t)b * NC + c) * DIN + d] = m;
}

__global__ void k_carry(const float* __restrict__ cmax, float* __restrict__ carry) {
    const int b = blockIdx.x, d = threadIdx.x;
    float run = -INFINITY;
    for (int c = NC - 1; c >= 0; --c) {
        const size_t idx = ((size_t)b * NC + c) * DIN + d;
        carry[idx] = run;                 // suffix-max of chunks strictly after c
        run = fmaxf(run, cmax[idx]);
    }
}

__global__ void k_suffix(const float* __restrict__ P, const float* __restrict__ carry,
                         uint16_t* __restrict__ S) {
    const int b = blockIdx.x, c = blockIdx.y, d = threadIdx.x;
    float run = carry[((size_t)b * NC + c) * DIN + d];
    const float*    p = P + ((size_t)b * T_ + (size_t)c * TC) * DIN + d;
    uint16_t*       s = S + ((size_t)b * T_ + (size_t)c * TC) * DIN + d;
    for (int i = TC - 1; i >= 0; --i) {
        run = fmaxf(run, p[(size_t)i * DIN]);
        s[(size_t)i * DIN] = f2bf(run);
    }
}

// ---------------- fused 4-layer MLP via bf16 WMMA ----------------

// One GEMM layer for a wave's 16 rows: acc[nt] = bias + A(16xK) * W(KxN_tile)
// Weights fragment-major: dword index ((kc*NT + nt)*32 + lane)*8 + j.
// Per-lane base pointers are hoisted so every DS access folds its constant
// offset (< 64 KB) into the 16-bit DS offset field -> no per-fragment VALU.
template <int NT, int KT>
__device__ __forceinline__ void gemm_layer(const uint16_t* __restrict__ act,
                                           const uint32_t* __restrict__ W,
                                           const float* __restrict__ bias,
                                           int lane, v8f acc[NT]) {
    const int m16 = lane & 15;
    const int hi  = lane >> 4;
    const int n16 = lane & 15;

    const uint32_t* wl = W + (size_t)lane * 8;         // per-lane weight base
    const uint16_t* al = act + m16 * SA + 8 * hi;      // per-lane A-row base (16B aligned)

#pragma unroll
    for (int nt = 0; nt < NT; ++nt) {
        const float bv = bias[nt * 16 + n16];
        v8f a0;
#pragma unroll
        for (int i = 0; i < 8; ++i) a0[i] = bv;
        acc[nt] = a0;
    }

#pragma unroll
    for (int kc = 0; kc < KT; ++kc) {
        // A fragment: 16x32 bf16 ISA layout; u[0..3] / u[4..7] contiguous -> 2 x ds_load_b128
        Frag a;
        a.q[0] = *(const uint4*)(al + kc * 32);
        a.q[1] = *(const uint4*)(al + kc * 32 + 16);
#pragma unroll
        for (int nt = 0; nt < NT; ++nt) {
            // B fragment: 8 contiguous dwords per lane -> 2 x ds_load_b128, imm offsets
            Frag b;
            const uint4* wp = (const uint4*)(wl + (size_t)(kc * NT + nt) * 256);
            b.q[0] = wp[0];
            b.q[1] = wp[1];
            acc[nt] = __builtin_amdgcn_wmma_f32_16x16x32_bf16(
                false, a.v, false, b.v, (short)0, acc[nt], false, false);
        }
    }
}

template <int NT>
__device__ __forceinline__ void store_hidden(uint16_t* __restrict__ act, int lane, v8f acc[NT]) {
    const int n16 = lane & 15;
    const int hi  = lane >> 4;
    uint16_t* al = act + 8 * hi * SA + n16;            // hoisted; (v*SA + nt*16) folds to imm
#pragma unroll
    for (int nt = 0; nt < NT; ++nt) {
#pragma unroll
        for (int v = 0; v < 8; ++v) {
            float x = fmaxf(acc[nt][v], 0.0f);                  // ReLU
            al[v * SA + nt * 16] = f2bf(x);                     // C/D layout -> row-major bf16
        }
    }
}

// Scatter one weight matrix (K x N, f32, row-major) into fragment-major bf16 LDS.
template <int NT, int K, int N>
__device__ __forceinline__ void fill_weights(uint16_t* __restrict__ hW,
                                             const float* __restrict__ Wg, int tid) {
    for (int i = tid; i < K * N; i += 256) {
        const int k = i / N, n = i % N;                // constant N -> mul-shift
        const int kc = k >> 5, kk = k & 31;
        const int hi = (kk >> 3) & 1;
        const int j  = ((kk & 7) >> 1) + 4 * (kk >> 4);
        const int lane = (n & 15) + 16 * hi;
        const int nt = n >> 4;
        hW[(((size_t)(kc * NT + nt) * 32 + lane) * 8 + j) * 2 + (kk & 1)] = f2bf(Wg[i]);
    }
}

__launch_bounds__(256, 2)
__global__ void k_mlp(const uint16_t* __restrict__ S,
                      const float* __restrict__ W1, const float* __restrict__ b1,
                      const float* __restrict__ W2, const float* __restrict__ b2,
                      const float* __restrict__ W3, const float* __restrict__ b3,
                      const float* __restrict__ W4, const float* __restrict__ b4,
                      float* __restrict__ out) {
    __shared__ uint32_t sW1[W1_DW];
    __shared__ uint32_t sW23[W23_DW];            // W2, then refilled with W3
    __shared__ uint32_t sW4[W4_DW];
    __shared__ float    sB[3 * 144 + 64];
    __shared__ uint16_t sAct[128 * SA];          // 128 rows x 168 bf16
    // total LDS ~= 148.4 KB -> 2 blocks (16 waves) per WGP

    const int tid = threadIdx.x;

    // --- zero weight/bias LDS (padding regions must be exact zero for WMMA) ---
    for (int i = tid; i < W1_DW; i += 256) sW1[i] = 0u;
    for (int i = tid; i < W23_DW; i += 256) sW23[i] = 0u;
    for (int i = tid; i < W4_DW; i += 256) sW4[i] = 0u;
    for (int i = tid; i < 3 * 144 + 64; i += 256) sB[i] = 0.0f;
    __syncthreads();

    // --- fill weights (f32 -> bf16, fragment-major) & biases ---
    fill_weights<9, DIN,  HID_>((uint16_t*)sW1,  W1, tid);
    fill_weights<9, HID_, HID_>((uint16_t*)sW23, W2, tid);
    fill_weights<4, HID_, DOUT>((uint16_t*)sW4,  W4, tid);
    for (int i = tid; i < HID_; i += 256) {
        sB[i] = b1[i]; sB[144 + i] = b2[i]; sB[288 + i] = b3[i];
    }
    if (tid < DOUT) sB[432 + tid] = b4[tid];

    // --- stage this block's 128 rows of S (bf16) into LDS, zero K-pad cols 144..159 ---
    const size_t rowBase = (size_t)blockIdx.x * 128;
    {
        const int row = tid >> 1, half = tid & 1;
        const uint4* src = (const uint4*)(S + (rowBase + row) * DIN + half * 64);
        uint4*       dst = (uint4*)(sAct + row * SA + half * 64);
#pragma unroll
        for (int i = 0; i < 8; ++i) dst[i] = src[i];
        if (half == 0) {
            uint4 z; z.x = z.y = z.z = z.w = 0u;
            uint4* zp = (uint4*)(sAct + row * SA + 144);
            zp[0] = z; zp[1] = z;
        }
    }
    __syncthreads();

    // --- per-wave fused MLP over its private 16 rows ---
    const int lane = tid & 31;
    const int wave = tid >> 5;
    uint16_t* act = sAct + wave * 16 * SA;

    v8f acc[9];
    gemm_layer<9, 4>(act, sW1, sB, lane, acc);          // 128 -> 144(132)
    store_hidden<9>(act, lane, acc);
    __syncthreads();
    gemm_layer<9, 5>(act, sW23, sB + 144, lane, acc);   // layer 2 (W2)
    store_hidden<9>(act, lane, acc);
    __syncthreads();
    fill_weights<9, HID_, HID_>((uint16_t*)sW23, W3, tid);  // overwrite same valid slots
    __syncthreads();
    gemm_layer<9, 5>(act, sW23, sB + 288, lane, acc);   // layer 3 (W3)
    store_hidden<9>(act, lane, acc);
    __syncthreads();

    v8f acc4[4];
    gemm_layer<4, 5>(act, sW4, sB + 432, lane, acc4);   // 160(132) -> 64

    const int n16 = lane & 15;
    const int hi  = lane >> 4;
    float* o = out + (rowBase + (size_t)wave * 16 + (size_t)8 * hi) * DOUT + n16;
#pragma unroll
    for (int nt = 0; nt < 4; ++nt) {
#pragma unroll
        for (int v = 0; v < 8; ++v) {
            const float x = acc4[nt][v];
            o[(size_t)v * DOUT + nt * 16] = 1.0f / (1.0f + __expf(-5.0f * x));
        }
    }
}

// ---------------- launcher ----------------

extern "C" void kernel_launch(void* const* d_in, const int* in_sizes, int n_in,
                              void* d_out, int out_size, void* d_ws, size_t ws_size,
                              hipStream_t stream) {
    const float* P  = (const float*)d_in[0];
    const float* b1 = (const float*)d_in[2];
    const float* W1 = (const float*)d_in[1];
    const float* W2 = (const float*)d_in[3];
    const float* b2 = (const float*)d_in[4];
    const float* W3 = (const float*)d_in[5];
    const float* b3 = (const float*)d_in[6];
    const float* W4 = (const float*)d_in[7];
    const float* b4 = (const float*)d_in[8];
    float* out = (float*)d_out;

    // workspace layout: S (bf16, 64 MiB) | chunkmax (1 MiB) | carry (1 MiB)
    uint16_t* S     = (uint16_t*)d_ws;
    float*    cmax  = (float*)((char*)d_ws + (size_t)M_ * DIN * sizeof(uint16_t));
    float*    carry = cmax + (size_t)B_ * NC * DIN;

    k_chunkmax<<<dim3(B_, NC), DIN, 0, stream>>>(P, cmax);
    k_carry<<<B_, DIN, 0, stream>>>(cmax, carry);
    k_suffix<<<dim3(B_, NC), DIN, 0, stream>>>(P, carry, S);
    k_mlp<<<M_ / 128, 256, 0, stream>>>(S, W1, b1, W2, b2, W3, b3, W4, b4, out);
}
